// Model_71691594105248
// MI455X (gfx1250) — matrix operand order repaired and checked by reading, not compiled, not run on hardware
//
#include <hip/hip_runtime.h>
#include <hip/hip_bf16.h>
#include <math.h>

typedef __attribute__((ext_vector_type(16))) _Float16 v16h;
typedef __attribute__((ext_vector_type(8)))  _Float16 v8h;
typedef __attribute__((ext_vector_type(8)))  float    v8f;

#define BB   32
#define SS   256
#define HH   768
#define NHD  12
#define DHD  64
#define FFD  3072
#define NC   9
#define TOK  (BB*SS)

#define BK   32
#define LDSK 40   // BK + 8 halves pad -> 80B row stride; all b128 LDS accesses stay 16B aligned

// LDS byte offset of a generic pointer into __shared__: flat LDS aperture keeps the
// workgroup-relative offset in addr[31:0] (ISA 10.2 aperture mapping).
__device__ __forceinline__ unsigned lds_off_of(const void* p) {
  return (unsigned)(unsigned long long)p;
}

// ---------------------------------------------------------------------------
// f16 WMMA GEMM with async-DMA tile staging.
//   A: f16 [M,K] row-major (lda), B: f16 [N,K] row-major (ldb)  -> C: f16 (ldc)
//   Block tile TBM x TBN, 8 waves, each wave owns 32 x 64 (2 x 4 WMMA tiles).
//   EPI: 0 = +bias(opt), 1 = +bias then exact GELU, 2 = *0.125 + attention-mask bias
//   Batched over gridDim.z with (b,h) = (z/nh, z%nh) strides (element units).
// ---------------------------------------------------------------------------
template<int TBM, int TBN, int EPI>
__global__ __launch_bounds__(256) void gemm_f16(
    const _Float16* __restrict__ A, const _Float16* __restrict__ Bm,
    const float* __restrict__ bias, _Float16* __restrict__ C,
    int K, int lda, int ldb, int ldc,
    long sAb, long sAh, long sBb, long sBh, long sCb, long sCh, int nh,
    const int* __restrict__ mask, int S)
{
  constexpr int WN  = TBN / 64;          // waves along N
  constexpr int NCA = TBM / 64;          // async b128 chunks per thread for A
  constexpr int NCB = TBN / 64;          // async b128 chunks per thread for B
  __shared__ __align__(16) _Float16 smem[(TBM + TBN) * LDSK];
  _Float16* ldsA = smem;
  _Float16* ldsB = smem + TBM * LDSK;

  const int tid  = threadIdx.x;
  const int lane = tid & 31;             // wave32
  const int wave = tid >> 5;
  const int bz = blockIdx.z / nh, hz = blockIdx.z % nh;
  const int bm0 = blockIdx.y * TBM;
  const int bn0 = blockIdx.x * TBN;
  const unsigned long long baseA =
      (unsigned long long)(A + (long)bz * sAb + (long)hz * sAh);
  const unsigned long long baseB =
      (unsigned long long)(Bm + (long)bz * sBb + (long)hz * sBh);
  const long offC = (long)bz * sCb + (long)hz * sCh;

  const int hi  = lane >> 4;             // half-wave select
  const int l15 = lane & 15;
  const int wm  = wave / WN, wn = wave % WN;

  // Precompute per-thread async copy descriptors (8-half = 16B chunks).
  unsigned aLds[NCA], aEl[NCA], bLds[NCB], bEl[NCB];
  #pragma unroll
  for (int i = 0; i < NCA; ++i) {
    int c = tid + i * 256;               // over TBM*4 chunks
    int m = c >> 2, kc = c & 3;
    aLds[i] = lds_off_of(ldsA + m * LDSK + kc * 8);
    aEl[i]  = (unsigned)((bm0 + m) * lda + kc * 8);
  }
  #pragma unroll
  for (int i = 0; i < NCB; ++i) {
    int c = tid + i * 256;               // over TBN*4 chunks
    int n = c >> 2, kc = c & 3;
    bLds[i] = lds_off_of(ldsB + n * LDSK + kc * 8);
    bEl[i]  = (unsigned)((bn0 + n) * ldb + kc * 8);
  }

  v8f acc[2][4] = {};

  for (int kt = 0; kt < K; kt += BK) {
    __syncthreads();                      // previous iteration's LDS reads done
    #pragma unroll
    for (int i = 0; i < NCA; ++i) {
      unsigned go = (aEl[i] + (unsigned)kt) * 2u;
      asm volatile("global_load_async_to_lds_b128 %0, %1, %2"
                   :: "v"(aLds[i]), "v"(go), "s"(baseA) : "memory");
    }
    #pragma unroll
    for (int i = 0; i < NCB; ++i) {
      unsigned go = (bEl[i] + (unsigned)kt) * 2u;
      asm volatile("global_load_async_to_lds_b128 %0, %1, %2"
                   :: "v"(bLds[i]), "v"(go), "s"(baseB) : "memory");
    }
    asm volatile("s_wait_asynccnt 0" ::: "memory");
    __syncthreads();                      // tile visible to all waves

    // A fragments (16x32 layout): lane<16 -> K {0..7,16..23}; lanes>=16 -> +8
    v16h af[2];
    #pragma unroll
    for (int mi = 0; mi < 2; ++mi) {
      const _Float16* pa = ldsA + (wm * 32 + mi * 16 + l15) * LDSK + hi * 8;
      v8h alo = *(const v8h*)(pa);
      v8h ahi = *(const v8h*)(pa + 16);
      af[mi] = __builtin_shufflevector(alo, ahi, 0,1,2,3,4,5,6,7,8,9,10,11,12,13,14,15);
    }
    #pragma unroll
    for (int j = 0; j < 4; ++j) {
      // B fragment (32x16): lane = N; lanes<16 -> K {0..7,16..23}; lanes>=16 -> +8
      const _Float16* pb = ldsB + (wn * 64 + j * 16 + l15) * LDSK + hi * 8;
      v8h blo = *(const v8h*)(pb);
      v8h bhi = *(const v8h*)(pb + 16);
      v16h bfr = __builtin_shufflevector(blo, bhi, 0,1,2,3,4,5,6,7,8,9,10,11,12,13,14,15);
      #pragma unroll
      for (int mi = 0; mi < 2; ++mi)
        acc[mi][j] = __builtin_amdgcn_wmma_f32_16x16x32_f16(
            false, af[mi], false, bfr, (short)0, acc[mi][j], false, false);
    }
  }

  // Epilogue: D VGPR r holds (M = r + 8*hi, N = lane&15)
  #pragma unroll
  for (int j = 0; j < 4; ++j) {
    const int cn = bn0 + wn * 64 + j * 16 + l15;
    float badd = 0.f;
    if (EPI != 2 && bias) badd = bias[cn];
    float mbias = 0.f;
    if (EPI == 2) mbias = (mask[(long)bz * S + cn] > 0) ? 0.f : -30000.f;
    #pragma unroll
    for (int mi = 0; mi < 2; ++mi) {
      #pragma unroll
      for (int r = 0; r < 8; ++r) {
        const int cm = bm0 + wm * 32 + mi * 16 + hi * 8 + r;
        float vv = acc[mi][j][r];
        if (EPI == 0)      { vv += badd; }
        else if (EPI == 1) { vv += badd; vv = 0.5f * vv * (1.f + erff(vv * 0.70710678118f)); }
        else               { vv = vv * 0.125f + mbias; }
        C[offC + (long)cm * ldc + cn] = (_Float16)vv;
      }
    }
  }
}

// ---------------------------------------------------------------------------
// Convert f32 weight [K,N] -> f16 transposed [N,K] (write-coalesced).
__global__ void convT_kernel(_Float16* __restrict__ out, const float* __restrict__ W,
                             int N, int K)
{
  const long gid = (long)blockIdx.x * 256 + threadIdx.x;
  if (gid >= (long)N * K) return;
  const int n = (int)(gid / K), k = (int)(gid % K);
  out[gid] = (_Float16)W[(long)k * N + n];
}

// v16 [B*S, 768] f16 -> vT16 [B*NH, DH, S] f16 (write-coalesced gather).
__global__ void vtrans_kernel(_Float16* __restrict__ vt, const _Float16* __restrict__ v)
{
  const long gid = (long)blockIdx.x * 256 + threadIdx.x;
  if (gid >= (long)BB * NHD * DHD * SS) return;
  const int s = (int)(gid & (SS - 1));
  const int d = (int)((gid >> 8) & (DHD - 1));
  const int z = (int)(gid >> 14);          // b*NH + h
  const int b = z / NHD, h = z % NHD;
  vt[gid] = v[((long)(b * SS + s)) * HH + h * DHD + d];
}

__global__ __launch_bounds__(256) void embed_kernel(
    _Float16* __restrict__ out, const int* __restrict__ x,
    const float* __restrict__ we, const float* __restrict__ pe,
    const float* __restrict__ te)
{
  const int token = blockIdx.x, tid = threadIdx.x;
  const int s = token & (SS - 1);
  const long wb = (long)x[token] * HH;
  const long base = (long)token * HH;
  #pragma unroll
  for (int i = 0; i < 3; ++i) {
    int idx = tid + i * 256;
    out[base + idx] = (_Float16)(we[wb + idx] + pe[(long)s * HH + idx] + te[idx]);
  }
}

// LayerNorm over H=768: in f16 (+ optional f32 residual) -> f32 and f16 outputs.
__global__ __launch_bounds__(256) void ln768(
    float* __restrict__ out32, _Float16* __restrict__ out16,
    const _Float16* __restrict__ in16, const float* __restrict__ res,
    const float* __restrict__ g, const float* __restrict__ b)
{
  __shared__ float red[256];
  const long base = (long)blockIdx.x * HH;
  const int tid = threadIdx.x;
  float v0 = (float)in16[base + tid];
  float v1 = (float)in16[base + tid + 256];
  float v2 = (float)in16[base + tid + 512];
  if (res) { v0 += res[base + tid]; v1 += res[base + tid + 256]; v2 += res[base + tid + 512]; }
  red[tid] = v0 + v1 + v2; __syncthreads();
  for (int s = 128; s > 0; s >>= 1) { if (tid < s) red[tid] += red[tid + s]; __syncthreads(); }
  const float mu = red[0] * (1.f / HH); __syncthreads();
  const float d0 = v0 - mu, d1 = v1 - mu, d2 = v2 - mu;
  red[tid] = d0 * d0 + d1 * d1 + d2 * d2; __syncthreads();
  for (int s = 128; s > 0; s >>= 1) { if (tid < s) red[tid] += red[tid + s]; __syncthreads(); }
  const float rs = rsqrtf(red[0] * (1.f / HH) + 1e-12f);
  const float y0 = d0 * rs * g[tid]       + b[tid];
  const float y1 = d1 * rs * g[tid + 256] + b[tid + 256];
  const float y2 = d2 * rs * g[tid + 512] + b[tid + 512];
  out32[base + tid]       = y0; out16[base + tid]       = (_Float16)y0;
  out32[base + tid + 256] = y1; out16[base + tid + 256] = (_Float16)y1;
  out32[base + tid + 512] = y2; out16[base + tid + 512] = (_Float16)y2;
}

// Row softmax (row length 256), f16 in-place, f32 math.
__global__ __launch_bounds__(256) void softmax256(_Float16* __restrict__ p)
{
  __shared__ float red[256];
  const long base = (long)blockIdx.x * 256;
  const int tid = threadIdx.x;
  const float v = (float)p[base + tid];
  red[tid] = v; __syncthreads();
  for (int s = 128; s > 0; s >>= 1) { if (tid < s) red[tid] = fmaxf(red[tid], red[tid + s]); __syncthreads(); }
  const float mx = red[0]; __syncthreads();
  const float e = __expf(v - mx);
  red[tid] = e; __syncthreads();
  for (int s = 128; s > 0; s >>= 1) { if (tid < s) red[tid] += red[tid + s]; __syncthreads(); }
  p[base + tid] = (_Float16)(e / red[0]);
}

// emissions[t,c] = h[t,:] . W_cls[:,c] + b_cls[c]   (N=9, too small for WMMA)
__global__ void cls_kernel(float* __restrict__ emis, const float* __restrict__ h,
                           const float* __restrict__ W, const float* __restrict__ bc)
{
  const long gid = (long)blockIdx.x * blockDim.x + threadIdx.x;
  if (gid >= (long)TOK * NC) return;
  const int t = (int)(gid / NC), c = (int)(gid % NC);
  const float* hr = h + (long)t * HH;
  float s = bc[c];
  for (int k = 0; k < HH; ++k) s += hr[k] * W[k * NC + c];
  emis[gid] = s;
}

// CRF NLL per batch element: forward logsumexp scan + gold-path score.
__global__ __launch_bounds__(32) void crf_kernel(
    const float* __restrict__ emis, const int* __restrict__ target,
    const float* __restrict__ start, const float* __restrict__ endt,
    const float* __restrict__ trans, float* __restrict__ llh)
{
  __shared__ float al[NC], tr[NC * NC];
  const int b = blockIdx.x, tid = threadIdx.x;
  for (int i = tid; i < NC * NC; i += 32) tr[i] = trans[i];
  const float* e = emis + (long)b * SS * NC;
  const int* tg = target + (long)b * SS;
  if (tid < NC) al[tid] = start[tid] + e[tid];
  __syncthreads();
  for (int t = 1; t < SS; ++t) {
    const bool valid = tg[t] > -1;
    float nv = 0.f;
    if (tid < NC) {
      float mx = -1e30f;
      for (int i = 0; i < NC; ++i) mx = fmaxf(mx, al[i] + tr[i * NC + tid]);
      float sm = 0.f;
      for (int i = 0; i < NC; ++i) sm += expf(al[i] + tr[i * NC + tid] - mx);
      nv = mx + logf(sm) + e[t * NC + tid];
    }
    __syncthreads();
    if (tid < NC && valid) al[tid] = nv;
    __syncthreads();
  }
  if (tid == 0) {
    float mx = -1e30f;
    for (int j = 0; j < NC; ++j) mx = fmaxf(mx, al[j] + endt[j]);
    float sm = 0.f;
    for (int j = 0; j < NC; ++j) sm += expf(al[j] + endt[j] - mx);
    const float den = mx + logf(sm);
    int prev = tg[0], last = tg[0];
    float num = start[prev] + e[prev];
    for (int t = 1; t < SS; ++t) {
      const int c = tg[t];
      if (c > -1) { num += tr[prev * NC + c] + e[t * NC + c]; prev = c; last = c; }
    }
    num += endt[last];
    llh[b] = num - den;
  }
}

__global__ void finalize_kernel(const float* __restrict__ llh, float* __restrict__ out)
{
  if (threadIdx.x == 0 && blockIdx.x == 0) {
    float s = 0.f;
    for (int i = 0; i < BB; ++i) s += llh[i];
    out[0] = -s / BB;
  }
}

// ---------------------------------------------------------------------------
extern "C" void kernel_launch(void* const* d_in, const int* in_sizes, int n_in,
                              void* d_out, int out_size, void* d_ws, size_t ws_size,
                              hipStream_t stream)
{
  (void)in_sizes; (void)n_in; (void)out_size; (void)ws_size;

  const int* x      = (const int*)d_in[0];
  const int* target = (const int*)d_in[1];
  const int* amask  = (const int*)d_in[2];
  const float* we   = (const float*)d_in[3];
  const float* pe   = (const float*)d_in[4];
  const float* te   = (const float*)d_in[5];
  const float* elng = (const float*)d_in[6];
  const float* elnb = (const float*)d_in[7];
  const float* Wcls = (const float*)d_in[8 + 16 * 12 + 0];
  const float* bcls = (const float*)d_in[8 + 16 * 12 + 1];
  const float* strt = (const float*)d_in[8 + 16 * 12 + 2];
  const float* endt = (const float*)d_in[8 + 16 * 12 + 3];
  const float* trns = (const float*)d_in[8 + 16 * 12 + 4];

  const long HB = (long)TOK * HH;                 // 6,291,456
  float* hbuf = (float*)d_ws;
  float* emis = hbuf + HB;
  float* llh  = emis + (long)TOK * NC;
  _Float16* hbuf16 = (_Float16*)(llh + 32);
  _Float16* q16  = hbuf16 + HB;
  _Float16* k16  = q16 + HB;
  _Float16* v16  = k16 + HB;
  _Float16* vT16 = v16 + HB;
  _Float16* e16  = vT16 + HB;
  _Float16* big16 = e16 + HB;                     // scores/attn[384,256,256] == ff1[8192,3072]
  _Float16* wslab = big16 + (long)BB * NHD * SS * SS;
  _Float16* wqT = wslab;
  _Float16* wkT = wqT + (long)HH * HH;
  _Float16* wvT = wkT + (long)HH * HH;
  _Float16* woT = wvT + (long)HH * HH;
  _Float16* w1T = woT + (long)HH * HH;            // [FFD,HH]
  _Float16* w2T = w1T + (long)FFD * HH;           // [HH,FFD]

  const dim3 blk(256);
  const long zero6[6] = {0, 0, 0, 0, 0, 0};
  (void)zero6;

  // embeddings -> q16 (raw sum), LN -> hbuf / hbuf16
  embed_kernel<<<TOK, 256, 0, stream>>>(q16, x, we, pe, te);
  ln768<<<TOK, 256, 0, stream>>>(hbuf, hbuf16, q16, nullptr, elng, elnb);

  for (int l = 0; l < 12; ++l) {
    #define LP(i) ((const float*)d_in[8 + 16 * l + (i)])
    const float *Wq = LP(0), *bq = LP(1), *Wk = LP(2), *bk = LP(3),
                *Wv = LP(4), *bv = LP(5), *Wo = LP(6), *bo = LP(7),
                *g1 = LP(8), *bn1 = LP(9), *W1 = LP(10), *bf1 = LP(11),
                *W2 = LP(12), *bf2 = LP(13), *g2 = LP(14), *bn2 = LP(15);
    #undef LP

    // Per-layer weight convert + transpose to f16 [N,K]
    convT_kernel<<<(HH * HH + 255) / 256, blk, 0, stream>>>(wqT, Wq, HH, HH);
    convT_kernel<<<(HH * HH + 255) / 256, blk, 0, stream>>>(wkT, Wk, HH, HH);
    convT_kernel<<<(HH * HH + 255) / 256, blk, 0, stream>>>(wvT, Wv, HH, HH);
    convT_kernel<<<(HH * HH + 255) / 256, blk, 0, stream>>>(woT, Wo, HH, HH);
    convT_kernel<<<(FFD * HH + 255) / 256, blk, 0, stream>>>(w1T, W1, FFD, HH);
    convT_kernel<<<(HH * FFD + 255) / 256, blk, 0, stream>>>(w2T, W2, HH, FFD);

    // Q, K, V projections: [8192,768] x [768,768]^T(f16)
    gemm_f16<128, 128, 0><<<dim3(HH / 128, TOK / 128, 1), blk, 0, stream>>>(
        hbuf16, wqT, bq, q16, HH, HH, HH, HH, 0, 0, 0, 0, 0, 0, 1, nullptr, SS);
    gemm_f16<128, 128, 0><<<dim3(HH / 128, TOK / 128, 1), blk, 0, stream>>>(
        hbuf16, wkT, bk, k16, HH, HH, HH, HH, 0, 0, 0, 0, 0, 0, 1, nullptr, SS);
    gemm_f16<128, 128, 0><<<dim3(HH / 128, TOK / 128, 1), blk, 0, stream>>>(
        hbuf16, wvT, bv, v16, HH, HH, HH, HH, 0, 0, 0, 0, 0, 0, 1, nullptr, SS);
    vtrans_kernel<<<((long)BB * NHD * DHD * SS + 255) / 256, blk, 0, stream>>>(vT16, v16);

    // scores[z=(b,h)] = q . k^T * 0.125 + mask_bias  (M=N=256, K=64, z=384)
    gemm_f16<128, 128, 2><<<dim3(SS / 128, SS / 128, BB * NHD), blk, 0, stream>>>(
        q16, k16, nullptr, big16, DHD, HH, HH, SS,
        (long)SS * HH, (long)DHD, (long)SS * HH, (long)DHD,
        (long)NHD * SS * SS, (long)SS * SS, NHD, amask, SS);
    softmax256<<<BB * NHD * SS, 256, 0, stream>>>(big16);

    // ctx[z] = attn . v  (M=256, N=64, K=256) -> e16 as [B,S,NH*DH]
    gemm_f16<256, 64, 0><<<dim3(1, 1, BB * NHD), blk, 0, stream>>>(
        big16, vT16, nullptr, e16, SS, SS, SS, HH,
        (long)NHD * SS * SS, (long)SS * SS, (long)NHD * DHD * SS, (long)DHD * SS,
        (long)SS * HH, (long)DHD, NHD, nullptr, SS);

    // attention output projection -> q16, then h = LN(h + q)
    gemm_f16<128, 128, 0><<<dim3(HH / 128, TOK / 128, 1), blk, 0, stream>>>(
        e16, woT, bo, q16, HH, HH, HH, HH, 0, 0, 0, 0, 0, 0, 1, nullptr, SS);
    ln768<<<TOK, 256, 0, stream>>>(hbuf, hbuf16, q16, hbuf, g1, bn1);

    // FFN: gelu(h W1 + b1) -> big16 ; big16 W2 + b2 -> q16 ; h = LN(h + q)
    gemm_f16<128, 128, 1><<<dim3(FFD / 128, TOK / 128, 1), blk, 0, stream>>>(
        hbuf16, w1T, bf1, big16, HH, HH, HH, FFD, 0, 0, 0, 0, 0, 0, 1, nullptr, SS);
    gemm_f16<128, 128, 0><<<dim3(HH / 128, TOK / 128, 1), blk, 0, stream>>>(
        big16, w2T, bf2, q16, FFD, FFD, FFD, HH, 0, 0, 0, 0, 0, 0, 1, nullptr, SS);
    ln768<<<TOK, 256, 0, stream>>>(hbuf, hbuf16, q16, hbuf, g2, bn2);
  }

  // classifier + CRF
  cls_kernel<<<((long)TOK * NC + 255) / 256, blk, 0, stream>>>(emis, hbuf, Wcls, bcls);
  crf_kernel<<<BB, 32, 0, stream>>>(emis, target, strt, endt, trns, llh);
  finalize_kernel<<<1, 32, 0, stream>>>(llh, (float*)d_out);
}
